// MetaKRec_51728586113410
// MI455X (gfx1250) — matrix operand order repaired and checked by reading, not compiled, not run on hardware
//
#include <hip/hip_runtime.h>
#include <hip/hip_bf16.h>

typedef _Float16 half_t;
typedef __attribute__((ext_vector_type(16))) _Float16 v16h;
typedef __attribute__((ext_vector_type(8)))  float    v8f;

static constexpr int DD = 64;   // feature dim (fixed by reference)

// ---------------------------------------------------------------- zero fill
__global__ void zero_kernel(float* __restrict__ p, long long n) {
    long long i = (long long)blockIdx.x * blockDim.x + threadIdx.x;
    if (i < n) p[i] = 0.0f;
}

// ---------------------------------------------------------------- degree (over dst)
__global__ void deg_kernel(const int* __restrict__ ei, float* __restrict__ deg, int E) {
    int e = blockIdx.x * blockDim.x + threadIdx.x;
    if (e >= E) return;
    int dst = ei[E + e];
    __hip_atomic_fetch_add(&deg[dst], 1.0f, __ATOMIC_RELAXED, __HIP_MEMORY_SCOPE_AGENT);
}

// ---------------------------------------------------------------- deg -> d^{-1/2} in place
__global__ void dinv_kernel(float* __restrict__ deg, int N) {
    int n = blockIdx.x * blockDim.x + threadIdx.x;
    if (n >= N) return;
    float d = deg[n];
    deg[n] = (d > 0.0f) ? rsqrtf(d) : 0.0f;
}

// ---------------------------------------------------------------- per-edge norm
__global__ void norm_kernel(const int* __restrict__ ei, const float* __restrict__ dinv,
                            float* __restrict__ nrm, int E) {
    int e = blockIdx.x * blockDim.x + threadIdx.x;
    if (e >= E) return;
    nrm[e] = dinv[ei[e]] * dinv[ei[E + e]];
}

// ---------------------------------------------------------------- scatter layer:
// hout[dst] += h[src] * norm.  16 lanes per edge, float4 per lane (64 floats).
__global__ void scatter_kernel(const float* __restrict__ hin, float* __restrict__ hout,
                               const int* __restrict__ ei, const float* __restrict__ nrm,
                               int E) {
    long long gid = (long long)blockIdx.x * blockDim.x + threadIdx.x;
    int e = (int)(gid >> 4);
    int t = (int)(gid & 15);
    if (e >= E) return;
    int   src = ei[e];
    int   dst = ei[E + e];
    float w   = nrm[e];
    const float4 v = *(const float4*)(hin + (size_t)src * DD + t * 4);
    float* op = hout + (size_t)dst * DD + t * 4;
    __hip_atomic_fetch_add(op + 0, v.x * w, __ATOMIC_RELAXED, __HIP_MEMORY_SCOPE_AGENT);
    __hip_atomic_fetch_add(op + 1, v.y * w, __ATOMIC_RELAXED, __HIP_MEMORY_SCOPE_AGENT);
    __hip_atomic_fetch_add(op + 2, v.z * w, __ATOMIC_RELAXED, __HIP_MEMORY_SCOPE_AGENT);
    __hip_atomic_fetch_add(op + 3, v.w * w, __ATOMIC_RELAXED, __HIP_MEMORY_SCOPE_AGENT);
}

// ---------------------------------------------------------------- Wh[k*64+j] = (f16) W[j*64+k]
__global__ void wprep_kernel(const float* __restrict__ W, half_t* __restrict__ Wh) {
    int idx = blockIdx.x * blockDim.x + threadIdx.x;
    if (idx >= DD * DD) return;
    int k = idx >> 6, j = idx & 63;
    Wh[idx] = (half_t)W[j * DD + k];
}

// ---------------------------------------------------------------- fused attention + node
// One wave32 per 16-node tile.  score = (emb @ W) @ a via v_wmma_f32_16x16x32_f16,
// then softmax over the 3 graph channels and weighted sum -> node[n, 0:64].
__global__ __launch_bounds__(256, 1)
void attn_node_kernel(const float* __restrict__ ls0, const float* __restrict__ ls1,
                      const float* __restrict__ ls2, const half_t* __restrict__ Wh,
                      const float* __restrict__ a_vec, float* __restrict__ node, int N) {
    const int lane   = threadIdx.x & 31;
    const int wave   = threadIdx.x >> 5;
    const int tile   = blockIdx.x * (blockDim.x >> 5) + wave;
    const int ntiles = (N + 15) >> 4;
    if (tile >= ntiles) return;
    const int node0  = tile << 4;

    const int half16 = lane >> 4;   // which 16-lane group
    const int sub    = lane & 15;
    const int koff   = half16 * 8;  // A-matrix K offset of this lane group is 2*koff

    // B fragments (W as 32x16 tiles).  B layout used here: lanes 0-15 hold K=0..15,
    // lanes 16-31 hold K=16..31, contiguous halves per lane.  Wh is [k][j] (K=j).
    v16h Bfrag[2][4];
#pragma unroll
    for (int ch = 0; ch < 2; ++ch)
#pragma unroll
        for (int cc = 0; cc < 4; ++cc) {
            const half_t* p = Wh + (size_t)(cc * 16 + sub) * DD + ch * 32 + half16 * 16;
            Bfrag[ch][cc] = *(const v16h*)p;
        }

    float aval[4];
#pragma unroll
    for (int cc = 0; cc < 4; ++cc) aval[cc] = a_vec[cc * 16 + sub];

    const float* ls[3] = {ls0, ls1, ls2};
    float sAcc[3][8];
#pragma unroll
    for (int g = 0; g < 3; ++g)
#pragma unroll
        for (int r = 0; r < 8; ++r) sAcc[g][r] = 0.0f;

#pragma unroll
    for (int g = 0; g < 3; ++g) {
        int rowN = node0 + sub; if (rowN >= N) rowN = N - 1;
        const float* rp = ls[g] + (size_t)rowN * DD;
        // A layout (16-bit 16x32): lanes 0-15: K=0..15; lanes 16-31: K=16..31 (matches B)
        v16h A[2];
#pragma unroll
        for (int ch = 0; ch < 2; ++ch) {
            const float* cp = rp + ch * 32;
            float4 g0a = *(const float4*)(cp + 2 * koff);
            float4 g0b = *(const float4*)(cp + 2 * koff + 4);
            float4 g1a = *(const float4*)(cp + 8 + 2 * koff);
            float4 g1b = *(const float4*)(cp + 12 + 2 * koff);
            v16h t;
            t[0]  = (half_t)g0a.x; t[1]  = (half_t)g0a.y; t[2]  = (half_t)g0a.z; t[3]  = (half_t)g0a.w;
            t[4]  = (half_t)g0b.x; t[5]  = (half_t)g0b.y; t[6]  = (half_t)g0b.z; t[7]  = (half_t)g0b.w;
            t[8]  = (half_t)g1a.x; t[9]  = (half_t)g1a.y; t[10] = (half_t)g1a.z; t[11] = (half_t)g1a.w;
            t[12] = (half_t)g1b.x; t[13] = (half_t)g1b.y; t[14] = (half_t)g1b.z; t[15] = (half_t)g1b.w;
            A[ch] = t;
        }
#pragma unroll
        for (int cc = 0; cc < 4; ++cc) {
            v8f C = {};
            C = __builtin_amdgcn_wmma_f32_16x16x32_f16(false, A[0], false, Bfrag[0][cc],
                                                       (short)0, C, false, false);
            C = __builtin_amdgcn_wmma_f32_16x16x32_f16(false, A[1], false, Bfrag[1][cc],
                                                       (short)0, C, false, false);
            // fold the "@ a" dot into the accumulator: lane holds column n = cc*16+sub
#pragma unroll
            for (int r = 0; r < 8; ++r) sAcc[g][r] += C[r] * aval[cc];
        }
    }

    // reduce the column dimension across each 16-lane group (rows m = r + 8*half16)
#pragma unroll
    for (int g = 0; g < 3; ++g)
#pragma unroll
        for (int r = 0; r < 8; ++r) {
#pragma unroll
            for (int m = 1; m <= 8; m <<= 1)
                sAcc[g][r] += __shfl_xor(sAcc[g][r], m, 32);
        }

    // softmax over the 3 channels + weighted sum.  Lane covers 2 feature columns.
    const int c0 = lane * 2;
#pragma unroll
    for (int n = 0; n < 16; ++n) {
        const int src = (n < 8) ? 0 : 16;
        const int r   = n & 7;
        float s0 = __shfl(sAcc[0][r], src, 32);
        float s1 = __shfl(sAcc[1][r], src, 32);
        float s2 = __shfl(sAcc[2][r], src, 32);
        float mx = fmaxf(s0, fmaxf(s1, s2));
        float e0 = __expf(s0 - mx), e1 = __expf(s1 - mx), e2 = __expf(s2 - mx);
        float inv = 1.0f / (e0 + e1 + e2);
        float w0 = e0 * inv, w1 = e1 * inv, w2 = e2 * inv;
        int nn = node0 + n;
        if (nn < N) {
            const float2 x0 = *(const float2*)(ls0 + (size_t)nn * DD + c0);
            const float2 x1 = *(const float2*)(ls1 + (size_t)nn * DD + c0);
            const float2 x2 = *(const float2*)(ls2 + (size_t)nn * DD + c0);
            float2 o;
            o.x = w0 * x0.x + w1 * x1.x + w2 * x2.x;
            o.y = w0 * x0.y + w1 * x1.y + w2 * x2.y;
            *(float2*)(node + (size_t)nn * DD + c0) = o;
        }
    }
}

// ---------------------------------------------------------------- out[b] = node[user[b]] . node[item[b]]
__global__ void dot_kernel(const int* __restrict__ user, const int* __restrict__ item,
                           const float* __restrict__ node, float* __restrict__ out, int B) {
    int gid  = blockIdx.x * blockDim.x + threadIdx.x;
    int b    = gid >> 5;
    int lane = threadIdx.x & 31;
    if (b >= B) return;
    int u = user[b], v = item[b];
    const float2 pa = *(const float2*)(node + (size_t)u * DD + lane * 2);
    const float2 pb = *(const float2*)(node + (size_t)v * DD + lane * 2);
    float s = pa.x * pb.x + pa.y * pb.y;
#pragma unroll
    for (int m = 16; m >= 1; m >>= 1) s += __shfl_xor(s, m, 32);
    if (lane == 0) out[b] = s;
}

// ================================================================ launch
extern "C" void kernel_launch(void* const* d_in, const int* in_sizes, int n_in,
                              void* d_out, int out_size, void* d_ws, size_t ws_size,
                              hipStream_t stream) {
    const int*   user = (const int*)d_in[0];
    const int*   item = (const int*)d_in[1];
    // d_in[2] = x = arange(N): emb_table[x] is the identity gather -> use emb_table directly
    const int*   ei[3] = {(const int*)d_in[3], (const int*)d_in[4], (const int*)d_in[5]};
    const float* emb  = (const float*)d_in[6];
    const float* W    = (const float*)d_in[7];
    const float* avec = (const float*)d_in[8];
    const int Bq = in_sizes[0];
    const int N  = in_sizes[2];
    const int E  = in_sizes[3] / 2;
    float* out = (float*)d_out;

    // carve workspace
    char*  base = (char*)d_ws;
    size_t off  = 0;
    auto carve = [&](size_t bytes) -> char* {
        char* p = base + off;
        off = (off + bytes + 255) & ~(size_t)255;
        return p;
    };
    float*  dinv  = (float*)carve((size_t)N * 4);
    float*  nrm   = (float*)carve((size_t)E * 4);
    float*  tmpA  = (float*)carve((size_t)N * DD * 4);
    float*  tmpB  = (float*)carve((size_t)N * DD * 4);
    float*  lsb[3];
    for (int g = 0; g < 3; ++g) lsb[g] = (float*)carve((size_t)N * DD * 4);
    float*  nodeb = (float*)carve((size_t)N * DD * 4);
    half_t* Wh    = (half_t*)carve((size_t)DD * DD * 2);

    const int TB = 256;
    wprep_kernel<<<(DD * DD + TB - 1) / TB, TB, 0, stream>>>(W, Wh);

    for (int g = 0; g < 3; ++g) {
        // degree / dinv / per-edge norm
        zero_kernel<<<(N + TB - 1) / TB, TB, 0, stream>>>(dinv, (long long)N);
        deg_kernel<<<(E + TB - 1) / TB, TB, 0, stream>>>(ei[g], dinv, E);
        dinv_kernel<<<(N + TB - 1) / TB, TB, 0, stream>>>(dinv, N);
        norm_kernel<<<(E + TB - 1) / TB, TB, 0, stream>>>(ei[g], dinv, nrm, E);

        // 3 propagation layers (layer 0 reads emb_table directly)
        const float* hin = emb;
        float* outs[3] = {tmpA, tmpB, lsb[g]};
        long long nh = (long long)N * DD;
        for (int l = 0; l < 3; ++l) {
            float* hout = outs[l];
            zero_kernel<<<(unsigned)((nh + TB - 1) / TB), TB, 0, stream>>>(hout, nh);
            long long tot = (long long)E * 16;
            scatter_kernel<<<(unsigned)((tot + TB - 1) / TB), TB, 0, stream>>>(hin, hout, ei[g], nrm, E);
            hin = hout;
        }
    }

    const int ntiles = (N + 15) / 16;
    const int wpb = 8;  // 8 wave32 per block, one 16-node tile per wave
    attn_node_kernel<<<(ntiles + wpb - 1) / wpb, wpb * 32, 0, stream>>>(
        lsb[0], lsb[1], lsb[2], Wh, avec, nodeb, N);

    dot_kernel<<<((long long)Bq * 32 + TB - 1) / TB, TB, 0, stream>>>(user, item, nodeb, out, Bq);
}
